// ODE_RNN_31301721653438
// MI455X (gfx1250) — compile-verified, matrix operand order checked by reading, not run on hardware
//
#include <hip/hip_runtime.h>
#include <hip/hip_bf16.h>

// ---------------------------------------------------------------------------
// ODE-RNN persistent kernel for MI455X (gfx1250), wave32 + WMMA bf16.
//   N_T=128, N_B=512, C=128, NHID=512, D_ODE=1024, N_RK4=8
// Each workgroup (1024 thr = 32 waves) owns a 16-row batch slice and runs the
// whole 128-step scan with LDS-resident state; batch rows are independent so
// no cross-WG synchronization is needed. All GEMMs use
// v_wmma_f32_16x16x32_bf16 (f32 accumulate); weights (~38 MB incl. bf16 x)
// are pre-converted once per launch and stay L2-resident (192 MB L2).
// ---------------------------------------------------------------------------

#define N_T   128
#define N_B   512
#define CCH   128
#define NHID  512
#define D_ODE 1024
#define KIC   16384          // N_T * CCH
#define KEXT  544            // 512 (h) + tb + s + 30 pad  (17 k-tiles of 32)

typedef __bf16 v16bf __attribute__((ext_vector_type(16)));
typedef __bf16 v8bf  __attribute__((ext_vector_type(8)));
typedef float  v8f   __attribute__((ext_vector_type(8)));

// -------------------- WMMA fragment helpers (wave32) -----------------------
// A fragment 16x32 bf16 (ISA 7.12.2): lane<16 -> row=lane, K: 0..7 (V0..3),
// 16..23 (V4..7); lane>=16 -> row=lane-16, K: 8..15 / 24..31. Per lane this is
// two 16-byte contiguous runs. B uses the identical layout from a pre-
// transposed [N][K] matrix (column-major B == row-major B^T).
__device__ __forceinline__ v16bf ld_frag_bf16(const __bf16* base, int ld, int lane) {
  int r = lane & 15, h = lane >> 4;
  const __bf16* p = base + (size_t)r * ld + h * 8;
  v8bf lo = *(const v8bf*)p;
  v8bf hi = *(const v8bf*)(p + 16);
  v16bf f;
#pragma unroll
  for (int i = 0; i < 8; ++i) { f[i] = lo[i]; f[i + 8] = hi[i]; }
  return f;
}

__device__ __forceinline__ v8f wmma_bf16(v16bf a, v16bf b, v8f c) {
  return __builtin_amdgcn_wmma_f32_16x16x32_bf16(false, a, false, b, (short)0, c,
                                                 false, false);
}

__device__ __forceinline__ float sigmoidf_(float x) {
  return 1.0f / (1.0f + __expf(-x));
}

// --------------------- weight / input prep (f32 -> bf16) -------------------
__global__ void prep_weights(const float* __restrict__ icW,   // [16384][512]
                             const float* __restrict__ Wih,   // [1536][128]
                             const float* __restrict__ Whh,   // [1536][512]
                             const float* __restrict__ W1,    // [514][1024]
                             const float* __restrict__ W2,    // [1024][512]
                             const float* __restrict__ x,     // [128][512][128]
                             __bf16* __restrict__ icWt,       // [512][16384]
                             __bf16* __restrict__ Wihb,       // [1536][128]
                             __bf16* __restrict__ Whhb,       // [1536][512]
                             __bf16* __restrict__ W1t,        // [1024][544]
                             __bf16* __restrict__ W2t,        // [512][1024]
                             __bf16* __restrict__ xb)         // [128][512][128]
{
  size_t idx = (size_t)blockIdx.x * blockDim.x + threadIdx.x;
  size_t stride = (size_t)gridDim.x * blockDim.x;
  for (size_t i = idx; i < (size_t)NHID * KIC; i += stride) {
    size_t n = i >> 14, k = i & (KIC - 1);
    icWt[i] = (__bf16)icW[k * NHID + n];
  }
  for (size_t i = idx; i < (size_t)3 * NHID * CCH; i += stride)
    Wihb[i] = (__bf16)Wih[i];
  for (size_t i = idx; i < (size_t)3 * NHID * NHID; i += stride)
    Whhb[i] = (__bf16)Whh[i];
  for (size_t i = idx; i < (size_t)D_ODE * KEXT; i += stride) {
    size_t n = i / KEXT, k = i % KEXT;
    W1t[i] = (k < NHID + 2) ? (__bf16)W1[k * D_ODE + n] : (__bf16)0.0f;
  }
  for (size_t i = idx; i < (size_t)NHID * D_ODE; i += stride) {
    size_t n = i >> 10, k = i & (D_ODE - 1);
    W2t[i] = (__bf16)W2[k * NHID + n];
  }
  for (size_t i = idx; i < (size_t)N_T * N_B * CCH; i += stride)
    xb[i] = (__bf16)x[i];
}

// ----------------------------- main persistent kernel ----------------------
__global__ __launch_bounds__(1024, 1)
void odernn_scan(const float* __restrict__ t_in,   // [N_T][N_B]
                 const float* __restrict__ ic_b,   // [NHID]
                 const float* __restrict__ b_ih,   // [3*NHID]
                 const float* __restrict__ b_hh,   // [3*NHID]
                 const float* __restrict__ ode_b1, // [D_ODE]
                 const float* __restrict__ ode_b2, // [NHID]
                 const __bf16* __restrict__ icWt,
                 const __bf16* __restrict__ Wihb,
                 const __bf16* __restrict__ Whhb,
                 const __bf16* __restrict__ W1t,
                 const __bf16* __restrict__ W2t,
                 const __bf16* __restrict__ xb,    // [N_T][N_B][C] bf16
                 float* __restrict__ out)          // [N_T+1][N_B][NHID]
{
  extern __shared__ char smem[];
  float*  hc   = (float*)smem;                          // [16][512]  32 KB
  float*  accK = (float*)(smem + 32768);                // [16][512]  32 KB
  __bf16* Ainp = (__bf16*)(smem + 65536);               // [16][544]  17 KB
  __bf16* z1t  = (__bf16*)(smem + 65536 + 17408);       // [16][1024] 32 KB

  const int tid  = threadIdx.x;
  const int wid  = tid >> 5;
  const int lane = tid & 31;
  const int b0   = blockIdx.x * 16;       // batch slice base row
  const int colw = wid * 16;              // this wave's column-tile base
  const int col  = colw + (lane & 15);    // output column owned by lane
  const int r0   = (lane >> 4) * 8;       // row base for C/D fragment
  const float dt = 1.0f / 8.0f;

  // zero the bf16 input tile (pad cols 514..543 stay zero forever)
  for (int i = tid; i < 16 * KEXT; i += blockDim.x) Ainp[i] = (__bf16)0.0f;
  __syncthreads();

  // ------------------- h0 = x_flat @ ic_W + ic_b ----------------------
  {
    v8f acc = {};
    const __bf16* bbase = icWt + (size_t)colw * KIC;
    for (int kt = 0; kt < KIC / 32; ++kt) {
      int tt = kt >> 2, cb = (kt & 3) * 32;   // x_flat[b, t*C + c] = x[t,b,c]
      v16bf a = ld_frag_bf16(xb + (size_t)tt * N_B * CCH + (size_t)b0 * CCH + cb,
                             CCH, lane);
      v16bf b = ld_frag_bf16(bbase + kt * 32, KIC, lane);
      if (kt + 2 < KIC / 32)   // cold streaming weights: prefetch 2 tiles ahead
        __builtin_prefetch(bbase + (size_t)(lane & 15) * KIC + (kt + 2) * 32, 0, 0);
      acc = wmma_bf16(a, b, acc);
    }
    float bias = ic_b[col];
#pragma unroll
    for (int v = 0; v < 8; ++v) {
      float hv = acc[v] + bias;
      hc[(r0 + v) * NHID + col]   = hv;
      Ainp[(r0 + v) * KEXT + col] = (__bf16)hv;       // bf16 h for GRU A
      out[(size_t)(b0 + r0 + v) * NHID + col] = hv;   // h_ode[0]
    }
  }
  __syncthreads();

  // --------------------------- time scan ------------------------------
  for (int ti = 0; ti < N_T; ++ti) {
    // ---- GRU cell: wave owns cols [colw, colw+16) of r/z/n gates -----
    {
      v8f aR = {}, aZ = {}, aIN = {}, aHN = {};
      const __bf16* xa = xb + (size_t)ti * N_B * CCH + (size_t)b0 * CCH;
#pragma unroll
      for (int kt = 0; kt < CCH / 32; ++kt) {                 // gi = x @ W_ih^T
        v16bf a  = ld_frag_bf16(xa + kt * 32, CCH, lane);
        v16bf bR = ld_frag_bf16(Wihb + (size_t)(colw)        * CCH + kt * 32, CCH, lane);
        v16bf bZ = ld_frag_bf16(Wihb + (size_t)(NHID + colw) * CCH + kt * 32, CCH, lane);
        v16bf bN = ld_frag_bf16(Wihb + (size_t)(2*NHID+colw) * CCH + kt * 32, CCH, lane);
        aR = wmma_bf16(a, bR, aR);
        aZ = wmma_bf16(a, bZ, aZ);
        aIN = wmma_bf16(a, bN, aIN);
      }
      for (int kt = 0; kt < NHID / 32; ++kt) {                // gh = h @ W_hh^T
        v16bf a  = ld_frag_bf16(Ainp + kt * 32, KEXT, lane);  // bf16 h in LDS
        v16bf bR = ld_frag_bf16(Whhb + (size_t)(colw)        * NHID + kt * 32, NHID, lane);
        v16bf bZ = ld_frag_bf16(Whhb + (size_t)(NHID + colw) * NHID + kt * 32, NHID, lane);
        v16bf bN = ld_frag_bf16(Whhb + (size_t)(2*NHID+colw) * NHID + kt * 32, NHID, lane);
        aR = wmma_bf16(a, bR, aR);
        aZ = wmma_bf16(a, bZ, aZ);
        aHN = wmma_bf16(a, bN, aHN);
      }
      // all waves must finish reading hc/Ainp before the epilogue writes them
      __syncthreads();
      float bR  = b_ih[col] + b_hh[col];
      float bZ  = b_ih[NHID + col] + b_hh[NHID + col];
      float bIN = b_ih[2 * NHID + col];
      float bHN = b_hh[2 * NHID + col];
#pragma unroll
      for (int v = 0; v < 8; ++v) {
        int r = r0 + v;
        float hprev = hc[r * NHID + col];
        float rg = sigmoidf_(aR[v] + bR);
        float zg = sigmoidf_(aZ[v] + bZ);
        float ng = tanhf(aIN[v] + bIN + rg * (aHN[v] + bHN));
        float hn = (1.0f - zg) * ng + zg * hprev;
        hc[r * NHID + col]   = hn;
        Ainp[r * KEXT + col] = (__bf16)hn;     // k1 input for first field eval
      }
    }
    if (tid < 16) {     // extended input columns: tb and s (s=0 at step start)
      Ainp[tid * KEXT + 512] = (__bf16)t_in[(size_t)ti * N_B + b0 + tid];
      Ainp[tid * KEXT + 513] = (__bf16)0.0f;
    }
    __syncthreads();

    // ---- RK4: 8 steps x 4 field evals, state fully in LDS ------------
    for (int step = 0; step < 8; ++step) {
      float s0 = (float)step * dt;
#pragma unroll 1
      for (int st = 0; st < 4; ++st) {
        // GEMM1: z1 = tanh(inp_ext @ W1_ext + b1); wave -> 2 N-tiles
        v8f a1 = {}, a2 = {};
        const int n2 = 512 + colw;
        for (int kt = 0; kt < KEXT / 32; ++kt) {
          v16bf a   = ld_frag_bf16(Ainp + kt * 32, KEXT, lane);
          v16bf b1f = ld_frag_bf16(W1t + (size_t)colw * KEXT + kt * 32, KEXT, lane);
          v16bf b2f = ld_frag_bf16(W1t + (size_t)n2   * KEXT + kt * 32, KEXT, lane);
          a1 = wmma_bf16(a, b1f, a1);
          a2 = wmma_bf16(a, b2f, a2);
        }
        {
          int c1 = colw + (lane & 15), c2 = n2 + (lane & 15);
          float bb1 = ode_b1[c1], bb2 = ode_b1[c2];
#pragma unroll
          for (int v = 0; v < 8; ++v) {
            z1t[(r0 + v) * D_ODE + c1] = (__bf16)tanhf(a1[v] + bb1);
            z1t[(r0 + v) * D_ODE + c2] = (__bf16)tanhf(a2[v] + bb2);
          }
        }
        __syncthreads();

        // GEMM2: k = z1 @ W2 + b2; wave owns cols [colw, colw+16) of NHID
        v8f acc = {};
        for (int kt = 0; kt < D_ODE / 32; ++kt) {
          v16bf a = ld_frag_bf16(z1t + kt * 32, D_ODE, lane);
          v16bf b = ld_frag_bf16(W2t + (size_t)colw * D_ODE + kt * 32, D_ODE, lane);
          acc = wmma_bf16(a, b, acc);
        }
        {
          float bb = ode_b2[col];
          float cnext = (st == 2) ? dt : 0.5f * dt;   // hc + cnext*k feeds next eval
#pragma unroll
          for (int v = 0; v < 8; ++v) {
            int r = r0 + v;
            float kv  = acc[v] + bb;
            float hcv = hc[r * NHID + col];
            if (st == 0)      accK[r * NHID + col] = kv;
            else if (st < 3)  accK[r * NHID + col] += 2.0f * kv;
            if (st < 3) {
              Ainp[r * KEXT + col] = (__bf16)(hcv + cnext * kv);
            } else {
              float sum = accK[r * NHID + col] + kv;       // k1+2k2+2k3+k4
              float hn = hcv + (dt / 6.0f) * sum;
              hc[r * NHID + col]   = hn;
              Ainp[r * KEXT + col] = (__bf16)hn;           // next step's k1 input
              if (step == 7)
                out[(size_t)(ti + 1) * N_B * NHID + (size_t)(b0 + r) * NHID + col] = hn;
            }
          }
        }
        if (tid < 16) {   // s for next field eval: {dt/2, dt/2, dt, dt}+s0
          float snext = (st < 2) ? (s0 + 0.5f * dt) : (s0 + dt);
          Ainp[tid * KEXT + 513] = (__bf16)snext;
        }
        __syncthreads();
      }
    }
  }
}

// ------------------------------ launcher -----------------------------------
extern "C" void kernel_launch(void* const* d_in, const int* in_sizes, int n_in,
                              void* d_out, int out_size, void* d_ws, size_t ws_size,
                              hipStream_t stream) {
  const float* t_in = (const float*)d_in[0];
  const float* x    = (const float*)d_in[1];
  const float* icW  = (const float*)d_in[2];
  const float* ic_b = (const float*)d_in[3];
  const float* W_ih = (const float*)d_in[4];
  const float* b_ih = (const float*)d_in[5];
  const float* W_hh = (const float*)d_in[6];
  const float* b_hh = (const float*)d_in[7];
  const float* W1   = (const float*)d_in[8];
  const float* b1   = (const float*)d_in[9];
  const float* W2   = (const float*)d_in[10];
  const float* b2   = (const float*)d_in[11];
  float* out = (float*)d_out;

  // bf16 weight + input workspace layout (~38 MB)
  char* ws = (char*)d_ws;
  __bf16* icWt = (__bf16*)(ws);                        // 512*16384*2  = 16777216
  __bf16* Wihb = (__bf16*)(ws + 16777216);             // 1536*128*2   =   393216
  __bf16* Whhb = (__bf16*)(ws + 17170432);             // 1536*512*2   =  1572864
  __bf16* W1t  = (__bf16*)(ws + 18743296);             // 1024*544*2   =  1114112
  __bf16* W2t  = (__bf16*)(ws + 19857408);             // 512*1024*2   =  1048576
  __bf16* xbf  = (__bf16*)(ws + 20905984);             // 128*512*128*2= 16777216

  prep_weights<<<2048, 256, 0, stream>>>(icW, W_ih, W_hh, W1, W2, x,
                                         icWt, Wihb, Whhb, W1t, W2t, xbf);

  const size_t smem = 32768u + 32768u + 17408u + 32768u;   // ~113 KB / WG
  odernn_scan<<<N_B / 16, 1024, smem, stream>>>(
      t_in, ic_b, b_ih, b_hh, b1, b2, icWt, Wihb, Whhb, W1t, W2t, xbf, out);
}
